// GraphAggregator_47519518163591
// MI455X (gfx1250) — matrix operand order repaired and checked by reading, not compiled, not run on hardware
//
#include <hip/hip_runtime.h>

typedef __bf16 bf16_t;
typedef __attribute__((ext_vector_type(16))) __bf16 v16bf;
typedef __attribute__((ext_vector_type(8)))  float  v8f;
typedef unsigned int uint;

__device__ __forceinline__ v8f wmma_bf16(v16bf a, v16bf b, v8f c) {
  return __builtin_amdgcn_wmma_f32_16x16x32_bf16(false, a, false, b, (short)0, c, false, false);
}

// A fragment (16x32 bf16, row-major source, K-contiguous). p = rowBase + g*8.
__device__ __forceinline__ v16bf load_a_frag(const __bf16* p) {
  v16bf a;
#pragma unroll
  for (int i = 0; i < 8; ++i) { a[i] = p[i]; a[8 + i] = p[16 + i]; }
  return a;
}
// B fragment (32x16 bf16) from K-major (transposed) source. p = colBase + g*8.
__device__ __forceinline__ v16bf load_b_frag(const __bf16* p) {
  v16bf b;
#pragma unroll
  for (int i = 0; i < 8; ++i) { b[i] = p[i]; b[8 + i] = p[16 + i]; }
  return b;
}
__device__ __forceinline__ v8f zero_v8f() {
  v8f z;
#pragma unroll
  for (int i = 0; i < 8; ++i) z[i] = 0.0f;
  return z;
}

// Generic shared-memory pointer -> 32-bit LDS byte address (ISA 10.2: for the
// LDS aperture, addr[31:0] is the wave-relative LDS address).
__device__ __forceinline__ uint lds_off(const void* p) {
  return (uint)(unsigned long long)p;
}
// CDNA5 async copy: global -> LDS, 16 bytes per lane, tracked by ASYNCcnt.
__device__ __forceinline__ void async_copy_b128(uint lds_byte, unsigned long long gaddr) {
  asm volatile("global_load_async_to_lds_b128 %0, %1, off"
               :: "v"(lds_byte), "v"(gaddr) : "memory");
}
__device__ __forceinline__ void async_wait0() {
  asm volatile("s_wait_asynccnt 0" ::: "memory");
}

// ---------------------------------------------------------------------------
// Kernel 1a: CPB table = relu(coords @ w1 + b1) @ w2   -> tbl[529][6]
// ---------------------------------------------------------------------------
__global__ __launch_bounds__(256) void cpb_table_kernel(
    const float* __restrict__ coords, const float* __restrict__ w1,
    const float* __restrict__ b1, const float* __restrict__ w2,
    float* __restrict__ tbl) {
  int e = blockIdx.x * 256 + threadIdx.x;
  if (e >= 529) return;
  float c0 = coords[e * 2 + 0];
  float c1 = coords[e * 2 + 1];
  float acc[6];
#pragma unroll
  for (int h = 0; h < 6; ++h) acc[h] = 0.0f;
  for (int j = 0; j < 512; ++j) {
    float hv = fmaxf(c0 * w1[j] + c1 * w1[512 + j] + b1[j], 0.0f);
#pragma unroll
    for (int h = 0; h < 6; ++h) acc[h] += hv * w2[j * 6 + h];
  }
#pragma unroll
  for (int h = 0; h < 6; ++h) tbl[e * 6 + h] = acc[h];
}

// ---------------------------------------------------------------------------
// Kernel 1b: bias[h][n][m] = 16*sigmoid(tbl[rel_idx[n][m]][h])
// ---------------------------------------------------------------------------
__global__ __launch_bounds__(256) void bias_kernel(
    const float* __restrict__ tbl, const int* __restrict__ rel_idx,
    float* __restrict__ bias) {
  int idx = blockIdx.x * 256 + threadIdx.x;     // 6*64*256
  if (idx >= 6 * 64 * 256) return;
  int h = idx / (64 * 256);
  int r = idx - h * (64 * 256);
  float v = tbl[rel_idx[r] * 6 + h];
  bias[idx] = 16.0f / (1.0f + __expf(-v));
}

// ---------------------------------------------------------------------------
// Generic 64-row WMMA GEMM:  C[64xN] = A[64xK] @ W[KxN] + bias
// MODE 0: A fp32  -> out bf16
// MODE 1: A fp32  -> per-32-column-chunk L2 normalize -> out bf16 (query path)
// MODE 2: A bf16 (async-DMA'd to LDS) -> out fp32 (final projection)
// LDS: As[64*K] bf16 (row major), Wt[N*K] bf16 (K-major / transposed),
//      MODE 1 additionally Cs[64*N] fp32.
// ---------------------------------------------------------------------------
template <int MODE>
__global__ __launch_bounds__(256) void gemm64_kernel(
    const float* __restrict__ Af, const bf16_t* __restrict__ Ab,
    const float* __restrict__ W, const float* __restrict__ bias,
    float* __restrict__ outf, bf16_t* __restrict__ outb, int N, int K) {
  extern __shared__ char smem[];
  bf16_t* As = (bf16_t*)smem;                 // 64*K
  bf16_t* Wt = As + 64 * K;                   // N*K (transposed)
  float*  Cs = (float*)(Wt + (size_t)N * K);  // MODE 1 only

  const int tid  = threadIdx.x;
  const int wave = tid >> 5;
  const int lane = tid & 31;
  const int g    = lane >> 4;
  const int l16  = lane & 15;
  const long m0  = (long)blockIdx.x * 64;

  // Warm L2 for the strided weight-transpose reads.
  for (int k = tid; k < K; k += 256) __builtin_prefetch(W + (size_t)k * N, 0, 0);

  if (MODE == 2) {
    // Async DMA the bf16 A tile into LDS; overlaps with the Wt conversion below.
    const char* src = (const char*)(Ab + m0 * K);
    const uint ldsA = lds_off(As);
    for (int idx = tid; idx < 8 * K; idx += 256)   // 64*K*2 bytes / 16
      async_copy_b128(ldsA + (uint)idx * 16u,
                      (unsigned long long)(src + (size_t)idx * 16));
  } else {
    const float* s = Af + m0 * K;
    for (int idx = tid; idx < 64 * K; idx += 256) As[idx] = (__bf16)s[idx];
  }
  // Weights transposed into LDS (K-contiguous per output column); L2-hot.
  for (int idx = tid; idx < N * K; idx += 256) {
    int n = idx / K;
    int k = idx - n * K;
    Wt[idx] = (__bf16)W[(long)k * N + n];
  }
  if (MODE == 2) async_wait0();
  __syncthreads();

  const int ntiles = 4 * (N >> 4);
  for (int t = wave; t < ntiles; t += 8) {
    int tm = t & 3;
    int tn = t >> 2;
    int arow = tm * 16 + l16;
    int col  = tn * 16 + l16;
    v8f acc = zero_v8f();
    for (int kb = 0; kb < K; kb += 32) {
      v16bf a  = load_a_frag(As + arow * K + kb + g * 8);
      v16bf bb = load_b_frag(Wt + (size_t)col * K + kb + g * 8);
      acc = wmma_bf16(a, bb, acc);
    }
    float bv = bias[col];
#pragma unroll
    for (int v = 0; v < 8; ++v) {
      int row = tm * 16 + g * 8 + v;
      float val = acc[v] + bv;
      if (MODE == 0)      outb[(m0 + row) * N + col] = (__bf16)val;
      else if (MODE == 2) outf[(m0 + row) * N + col] = val;
      else                Cs[row * N + col] = val;
    }
  }

  if (MODE == 1) {
    __syncthreads();
    int chunks = N >> 5;  // 32-wide head chunks
    for (int idx = tid; idx < 64 * chunks; idx += 256) {
      int r  = idx / chunks;
      int hh = idx - r * chunks;
      const float* p = Cs + r * N + hh * 32;
      float s = 0.0f;
#pragma unroll
      for (int i = 0; i < 32; ++i) s += p[i] * p[i];
      float inv = 1.0f / fmaxf(sqrtf(s), 1e-12f);
      bf16_t* o = outb + (m0 + r) * N + hh * 32;
#pragma unroll
      for (int i = 0; i < 32; ++i) o[i] = (__bf16)(p[i] * inv);
    }
  }
}

// ---------------------------------------------------------------------------
// Kernel 4: fused per-window attention. One block per window (1152 blocks).
// LDS: qn[64*192]bf16 | kn[256*192]bf16 (normalized, m-major) |
//      ftT[192*256]bf16 (value, d-major) | sim[64*264]f32 | reductions.
// ---------------------------------------------------------------------------
__global__ __launch_bounds__(256) void attn_kernel(
    const bf16_t* __restrict__ qn_g, const bf16_t* __restrict__ xs_g,
    const float* __restrict__ graph, const float* __restrict__ logit_scale,
    const float* __restrict__ bias_g, bf16_t* __restrict__ out_g) {
  extern __shared__ char smem[];
  bf16_t* qn_s   = (bf16_t*)smem;            // 64*192
  bf16_t* kn_s   = qn_s + 64 * 192;          // 256*192
  bf16_t* ftT_s  = kn_s + 256 * 192;         // 192*256
  float*  sim_s  = (float*)(ftT_s + 192 * 256);  // 64*264 (padded rows)
  float*  red_s  = sim_s + 64 * 264;         // 256
  float*  rowm   = red_s + 256;              // 64
  float*  rowinv = rowm + 64;                // 64

  const int tid  = threadIdx.x;
  const int wave = tid >> 5;
  const int lane = tid & 31;
  const int g    = lane >> 4;
  const int l16  = lane & 15;

  const int wid = blockIdx.x;
  const int b   = wid / 576;
  const int rem = wid - b * 576;
  const int hi  = rem / 24;
  const int wi  = rem - hi * 24;

  // Phase A: gather normalized-q window rows via async global->LDS DMA.
  // 64 rows x 384 contiguous bytes = 1536 x 16B chunks (6 per thread).
  {
    const char* base = (const char*)qn_g;
    const uint ldsQ = lds_off(qn_s);
    for (int idx = tid; idx < 1536; idx += 256) {
      int r = idx / 24, ch = idx - r * 24;   // 24 x 16B chunks per row
      int y = hi * 8 + (r >> 3), x = wi * 8 + (r & 7);
      long p = (long)(b * 192 + y) * 192 + x;
      async_copy_b128(ldsQ + (uint)idx * 16u,
                      (unsigned long long)(base + p * 384 + ch * 16));
    }
  }
  // Phase B1: key half of xs neighborhood (zero-padded OOB -> VALU path).
  {
    uint* dst = (uint*)kn_s;
    const uint* src = (const uint*)xs_g;
    for (int idx = tid; idx < 256 * 96; idx += 256) {
      int m = idx / 96, dw = idx - m * 96;
      int sy = hi * 8 - 4 + (m >> 4);
      int sx = wi * 8 - 4 + (m & 15);
      uint v = 0u;
      if (sy >= 0 && sy < 192 && sx >= 0 && sx < 192)
        v = src[((long)(b * 192 + sy) * 192 + sx) * 192 + dw];
      dst[idx] = v;
    }
  }
  // Phase B2: value half, stored transposed (d-major) for B fragments.
  {
    const uint* src = (const uint*)xs_g;
    for (int idx = tid; idx < 256 * 96; idx += 256) {
      int m  = idx & 255;     // == tid for stride 256
      int cp = idx >> 8;      // dword pair -> columns 2cp, 2cp+1
      int sy = hi * 8 - 4 + (m >> 4);
      int sx = wi * 8 - 4 + (m & 15);
      uint v = 0u;
      if (sy >= 0 && sy < 192 && sx >= 0 && sx < 192)
        v = src[((long)(b * 192 + sy) * 192 + sx) * 192 + 96 + cp];
      union { uint u; __bf16 h[2]; } cv;
      cv.u = v;
      ftT_s[(2 * cp + 0) * 256 + m] = cv.h[0];
      ftT_s[(2 * cp + 1) * 256 + m] = cv.h[1];
    }
  }
  async_wait0();     // Phase A DMA complete (this wave) before barrier
  __syncthreads();

  // Phase B3: cosine-normalize keys per (row, head) in place.
  {
    int m = tid;  // 256 rows
#pragma unroll
    for (int h = 0; h < 6; ++h) {
      bf16_t* p = kn_s + m * 192 + h * 32;
      float s = 0.0f;
#pragma unroll
      for (int i = 0; i < 32; ++i) { float v = (float)p[i]; s += v * v; }
      float inv = 1.0f / fmaxf(sqrtf(s), 1e-12f);
#pragma unroll
      for (int i = 0; i < 32; ++i) p[i] = (__bf16)((float)p[i] * inv);
    }
  }
  __syncthreads();

  for (int h = 0; h < 6; ++h) {
    float scale = __expf(fminf(logit_scale[h], 4.6052f));

    // Phase C: sim = (qn @ kn^T)*graph*scale + bias  (64x256, one WMMA/tile).
    for (int t = wave; t < 64; t += 8) {
      int tm = t & 3, tn = t >> 2;
      v16bf a  = load_a_frag(qn_s + (tm * 16 + l16) * 192 + h * 32 + g * 8);
      int mcol = tn * 16 + l16;
      v16bf bb = load_b_frag(kn_s + mcol * 192 + h * 32 + g * 8);
      v8f acc = zero_v8f();
      acc = wmma_bf16(a, bb, acc);
#pragma unroll
      for (int v = 0; v < 8; ++v) {
        int row = tm * 16 + g * 8 + v;
        float gr  = graph[((long)wid * 64 + row) * 256 + mcol];
        float bsv = bias_g[((h * 64 + row) << 8) + mcol];
        sim_s[row * 264 + mcol] = acc[v] * gr * scale + bsv;
      }
    }
    __syncthreads();

    // Phase D: fp32 softmax over m=256 (4 threads per row).
    {
      int r = tid >> 2, qq = tid & 3;
      float* p = sim_s + r * 264 + qq * 64;
      float mx = -1e30f;
#pragma unroll 8
      for (int i = 0; i < 64; ++i) mx = fmaxf(mx, p[i]);
      red_s[tid] = mx;
      __syncthreads();
      if (tid < 64)
        rowm[tid] = fmaxf(fmaxf(red_s[tid * 4 + 0], red_s[tid * 4 + 1]),
                          fmaxf(red_s[tid * 4 + 2], red_s[tid * 4 + 3]));
      __syncthreads();
      float rm = rowm[r];
      float s = 0.0f;
#pragma unroll 8
      for (int i = 0; i < 64; ++i) {
        float e = __expf(p[i] - rm);
        p[i] = e;
        s += e;
      }
      red_s[tid] = s;
      __syncthreads();
      if (tid < 64)
        rowinv[tid] = 1.0f / (red_s[tid * 4 + 0] + red_s[tid * 4 + 1] +
                              red_s[tid * 4 + 2] + red_s[tid * 4 + 3]);
      __syncthreads();
    }

    // Phase E: out_h = attn @ feat_h (64x32), one 16x16 tile per wave.
    {
      int tm = wave >> 1, tn = wave & 1;
      int arow = tm * 16 + l16;
      float rinv = rowinv[arow];
      int ncol = tn * 16 + l16;
      const bf16_t* pbbase = ftT_s + (h * 32 + ncol) * 256;
      v8f acc = zero_v8f();
      for (int kb = 0; kb < 256; kb += 32) {
        const float* pa = sim_s + arow * 264 + kb + g * 8;
        v16bf a;
#pragma unroll
        for (int i = 0; i < 8; ++i) {
          a[i]     = (__bf16)(pa[i] * rinv);        // fold 1/sum here
          a[8 + i] = (__bf16)(pa[16 + i] * rinv);
        }
        v16bf bb = load_b_frag(pbbase + kb + g * 8);
        acc = wmma_bf16(a, bb, acc);
      }
#pragma unroll
      for (int v = 0; v < 8; ++v) {
        int row = tm * 16 + g * 8 + v;
        out_g[((long)wid * 64 + row) * 192 + h * 32 + ncol] = (__bf16)acc[v];
      }
    }
    __syncthreads();
  }
}

// ---------------------------------------------------------------------------
// Launch
// ---------------------------------------------------------------------------
extern "C" void kernel_launch(void* const* d_in, const int* in_sizes, int n_in,
                              void* d_out, int out_size, void* d_ws,
                              size_t ws_size, hipStream_t stream) {
  (void)in_sizes; (void)n_in; (void)out_size; (void)ws_size;
  const float* x        = (const float*)d_in[0];
  const float* graph    = (const float*)d_in[1];
  const float* lscale   = (const float*)d_in[2];
  const float* w_group  = (const float*)d_in[3];
  const float* b_group  = (const float*)d_in[4];
  const float* w_sample = (const float*)d_in[5];
  const float* b_sample = (const float*)d_in[6];
  const float* w_proj   = (const float*)d_in[7];
  const float* b_proj   = (const float*)d_in[8];
  const float* cpb_w1   = (const float*)d_in[9];
  const float* cpb_b1   = (const float*)d_in[10];
  const float* cpb_w2   = (const float*)d_in[11];
  const float* coords   = (const float*)d_in[12];
  const int*   rel_idx  = (const int*)d_in[13];
  float* out = (float*)d_out;

  // Workspace layout (bytes). Total ~113.7 MB (bf16 staging; mostly L2-hot).
  char* ws = (char*)d_ws;
  bf16_t* xs_ws   = (bf16_t*)(ws + 0);                 // 73728*384 bf16
  bf16_t* qn_ws   = (bf16_t*)(ws + 56623104);          // 73728*192 bf16
  bf16_t* ao_ws   = (bf16_t*)(ws + 84934656);          // 73728*192 bf16
  float*  tbl_ws  = (float*)(ws + 113246208);          // 529*6 f32
  float*  bias_ws = (float*)(ws + 113246208 + 16384);  // 6*64*256 f32

  // 1) CPB table + relative-position bias (tiny).
  cpb_table_kernel<<<3, 256, 0, stream>>>(coords, cpb_w1, cpb_b1, cpb_w2, tbl_ws);
  bias_kernel<<<384, 256, 0, stream>>>(tbl_ws, rel_idx, bias_ws);

  // 2) xs = x @ w_sample + b_sample  (M=73728, N=384, K=192) -> bf16
  {
    size_t sh = (size_t)(64 * 192 + 384 * 192) * sizeof(bf16_t);  // 172032
    gemm64_kernel<0><<<1152, 256, sh, stream>>>(x, (const bf16_t*)nullptr,
                                                w_sample, b_sample, (float*)nullptr,
                                                xs_ws, 384, 192);
  }
  // 3) qn = normalize_per_head(x @ w_group + b_group) -> bf16
  {
    size_t sh = (size_t)(64 * 192 + 192 * 192) * sizeof(bf16_t) +
                (size_t)64 * 192 * sizeof(float);                 // 147456
    gemm64_kernel<1><<<1152, 256, sh, stream>>>(x, (const bf16_t*)nullptr,
                                                w_group, b_group, (float*)nullptr,
                                                qn_ws, 192, 192);
  }
  // 4) Fused attention per window.
  {
    size_t sh = (size_t)64 * 192 * 2 + (size_t)256 * 192 * 2 +
                (size_t)192 * 256 * 2 + (size_t)64 * 264 * 4 +
                (size_t)(256 + 64 + 64) * 4;                      // 290304
    attn_kernel<<<1152, 256, sh, stream>>>(qn_ws, xs_ws, graph, lscale,
                                           bias_ws, ao_ws);
  }
  // 5) out = attn_out @ w_proj + b_proj -> fp32 d_out
  {
    size_t sh = (size_t)(64 * 192 + 192 * 192) * sizeof(bf16_t);  // 98304
    gemm64_kernel<2><<<1152, 256, sh, stream>>>((const float*)nullptr, ao_ws,
                                                w_proj, b_proj, out,
                                                (bf16_t*)nullptr, 192, 192);
  }
}
